// TemporalFusionTransformer_6382321401904
// MI455X (gfx1250) — matrix operand order repaired and checked by reading, not compiled, not run on hardware
//
#include <hip/hip_runtime.h>
#include <hip/hip_bf16.h>

// ---------------------------------------------------------------------------
// CDNA5 (gfx1250) TFT forward. All GEMMs via v_wmma_f32_16x16x32_bf16 with
// bf16 operands resident in memory (no per-tile f32->bf16 conversion).
// wave32 everywhere; LN reductions via 32-lane shfl_xor.
// ---------------------------------------------------------------------------

typedef __attribute__((ext_vector_type(16))) __bf16 v16bf;
typedef __attribute__((ext_vector_type(8)))  __bf16 v8bf;
typedef __attribute__((ext_vector_type(8)))  float  v8f;

__device__ __forceinline__ __bf16 f2bf(float x) {
  union { float f; unsigned u; } a; a.f = x;
  unsigned r = a.u + 0x7FFFu + ((a.u >> 16) & 1u);   // RNE-ish
  unsigned short h = (unsigned short)(r >> 16);
  return __builtin_bit_cast(__bf16, h);
}
__device__ __forceinline__ float bf2f(__bf16 x) {
  unsigned u = (unsigned)__builtin_bit_cast(unsigned short, x) << 16;
  return __builtin_bit_cast(float, u);
}

__device__ __forceinline__ float sigm(float x) { return 1.f / (1.f + __expf(-x)); }

__device__ __forceinline__ float wave_sum(float v) {
  #pragma unroll
  for (int off = 16; off > 0; off >>= 1) v += __shfl_xor(v, off, 32);
  return v;
}
__device__ __forceinline__ float wave_max(float v) {
  #pragma unroll
  for (int off = 16; off > 0; off >>= 1) v = fmaxf(v, __shfl_xor(v, off, 32));
  return v;
}

__device__ __forceinline__ v16bf cat8(v8bf lo, v8bf hi) {
  return __builtin_shufflevector(lo, hi, 0, 1, 2, 3, 4, 5, 6, 7,
                                          8, 9, 10, 11, 12, 13, 14, 15);
}

// ---------------------------------------------------------------------------
// GEMM: C[M,N] = act(A[M,K] @ W[N,K]^T + bias[N]); A,W bf16; C fp32 and/or bf16
// One wave -> 16x32 tile (2 WMMAs/K-step, shared A fragment); 4 waves -> 16x128.
// K %32, M %16, lda/ldw even. Out-of-range columns clamped, never stored.
// act: 0=none, 1=ELU, 2=sigmoid
// ---------------------------------------------------------------------------
__global__ void gemm_bf16_wmma(const __bf16* __restrict__ A, int lda,
                               const __bf16* __restrict__ W, int ldw,
                               const float* __restrict__ bias,
                               float* __restrict__ C, __bf16* __restrict__ Cb,
                               int ldc, int M, int N, int K, int act) {
  const int lane = threadIdx.x & 31;
  const int wave = threadIdx.x >> 5;
  const int n0 = (blockIdx.x * 4 + wave) * 32;
  const int m0 = blockIdx.y * 16;
  const int lm = lane & 15;
  const int hi = lane >> 4;
  const int m  = m0 + lm;

  const int nc0 = n0 + lm;
  const int nc1 = n0 + 16 + lm;
  const int nc0c = (nc0 < N) ? nc0 : (N - 1);   // branchless clamp
  const int nc1c = (nc1 < N) ? nc1 : (N - 1);

  const __bf16* Arow = A + (size_t)m * lda;
  const __bf16* Wr0  = W + (size_t)nc0c * ldw;
  const __bf16* Wr1  = W + (size_t)nc1c * ldw;

  v8f acc0 = {}, acc1 = {};
  #pragma unroll 2
  for (int k0 = 0; k0 < K; k0 += 32) {
    // A fragment: lanes 0-15 K=k0..k0+15, lanes 16-31 K=k0+16..k0+31 (matches B)
    const __bf16* ap = Arow + k0 + hi * 16;
    v16bf af = cat8(*(const v8bf*)ap, *(const v8bf*)(ap + 8));

    // B fragments: lanes 0-15 K=k0..k0+15, lanes 16-31 K=k0+16..k0+31
    const __bf16* w0 = Wr0 + k0 + hi * 16;
    v16bf bf0 = cat8(*(const v8bf*)w0, *(const v8bf*)(w0 + 8));
    acc0 = __builtin_amdgcn_wmma_f32_16x16x32_bf16(false, af, false, bf0,
                                                   (short)0, acc0, false, false);

    const __bf16* w1 = Wr1 + k0 + hi * 16;
    v16bf bf1 = cat8(*(const v8bf*)w1, *(const v8bf*)(w1 + 8));
    acc1 = __builtin_amdgcn_wmma_f32_16x16x32_bf16(false, af, false, bf1,
                                                   (short)0, acc1, false, false);
  }

  // epilogue: lane holds cols nc0 / nc1, rows m0 + hi*8 + v
  if (nc0 < N) {
    float bb = bias ? bias[nc0] : 0.f;
    #pragma unroll
    for (int v = 0; v < 8; ++v) {
      int cm = m0 + hi * 8 + v;
      float val = acc0[v] + bb;
      if (act == 1)      val = (val > 0.f) ? val : (__expf(val) - 1.f);
      else if (act == 2) val = sigm(val);
      if (C)  C[(size_t)cm * ldc + nc0]  = val;
      if (Cb) Cb[(size_t)cm * ldc + nc0] = f2bf(val);
    }
  }
  if (nc1 < N) {
    float bb = bias ? bias[nc1] : 0.f;
    #pragma unroll
    for (int v = 0; v < 8; ++v) {
      int cm = m0 + hi * 8 + v;
      float val = acc1[v] + bb;
      if (act == 1)      val = (val > 0.f) ? val : (__expf(val) - 1.f);
      else if (act == 2) val = sigm(val);
      if (C)  C[(size_t)cm * ldc + nc1]  = val;
      if (Cb) Cb[(size_t)cm * ldc + nc1] = f2bf(val);
    }
  }
}

static inline void gemm(hipStream_t s, const __bf16* A, int lda, const __bf16* W, int ldw,
                        const float* bias, float* C, __bf16* Cb, int ldc,
                        int M, int N, int K, int act) {
  dim3 g((N + 127) / 128, (M + 15) / 16);
  gemm_bf16_wmma<<<g, dim3(128), 0, s>>>(A, lda, W, ldw, bias, C, Cb, ldc, M, N, K, act);
}

// --------------------------- elementwise kernels ---------------------------

__global__ void fill_zero_kernel(float* p, int n) {
  int i = blockIdx.x * blockDim.x + threadIdx.x;
  if (i < n) p[i] = 0.f;
}

__global__ void f2bf_kernel(const float* __restrict__ in, __bf16* __restrict__ out, int n) {
  int i = blockIdx.x * blockDim.x + threadIdx.x;
  if (i < n) out[i] = f2bf(in[i]);
}
static inline void conv(hipStream_t s, const float* src, __bf16* dst, int n) {
  f2bf_kernel<<<(n + 255) / 256, 256, 0, s>>>(src, dst, n);
}

// h1[n,i] = elu(x[n,f]*w[f,i] + b[f,i])  -> bf16
__global__ void vsn_h1_kernel(const float* __restrict__ x, const float* __restrict__ w,
                              const float* __restrict__ b, __bf16* __restrict__ h1,
                              int rows, int f) {
  int idx = blockIdx.x * blockDim.x + threadIdx.x;
  if (idx >= rows * 128) return;
  int n = idx >> 7, i = idx & 127;
  float v = x[(size_t)n * 9 + f] * w[f * 128 + i] + b[f * 128 + i];
  h1[idx] = f2bf((v > 0.f) ? v : (__expf(v) - 1.f));
}

// proc row = LN(x*skw+skb + h*g), per-feature gamma/beta; 1 wave/row -> bf16
__global__ void vsn_proc_kernel(const float* __restrict__ x,
                                const float* __restrict__ skw, const float* __restrict__ skb,
                                const float* __restrict__ h, const float* __restrict__ g,
                                const float* __restrict__ gam, const float* __restrict__ bet,
                                __bf16* __restrict__ out, int ostride, int rows, int f) {
  int wave = threadIdx.x >> 5, lane = threadIdx.x & 31;
  int n = blockIdx.x * (blockDim.x >> 5) + wave;
  if (n >= rows) return;
  float xv = x[(size_t)n * 9 + f];
  float v[4], su = 0.f, ss = 0.f;
  #pragma unroll
  for (int j = 0; j < 4; ++j) {
    int i = lane * 4 + j;
    float s = xv * skw[f * 128 + i] + skb[f * 128 + i];
    float val = s + h[(size_t)n * 128 + i] * g[(size_t)n * 128 + i];
    v[j] = val; su += val; ss += val * val;
  }
  su = wave_sum(su); ss = wave_sum(ss);
  float mean = su * (1.f / 128.f);
  float var = ss * (1.f / 128.f) - mean * mean;
  float inv = rsqrtf(var + 1e-5f);
  #pragma unroll
  for (int j = 0; j < 4; ++j) {
    int i = lane * 4 + j;
    out[(size_t)n * ostride + i] = f2bf((v[j] - mean) * inv * gam[f * 128 + i] + bet[f * 128 + i]);
  }
}

// out(bf16) = LN(s + h*g), rows of 128; 1 wave/row
__global__ void grn_finish_kernel(const float* __restrict__ s, const float* __restrict__ h,
                                  const float* __restrict__ g, const float* __restrict__ gam,
                                  const float* __restrict__ bet, __bf16* __restrict__ out,
                                  int rows) {
  int wave = threadIdx.x >> 5, lane = threadIdx.x & 31;
  int n = blockIdx.x * (blockDim.x >> 5) + wave;
  if (n >= rows) return;
  float v[4], su = 0.f, ss = 0.f;
  #pragma unroll
  for (int j = 0; j < 4; ++j) {
    size_t id = (size_t)n * 128 + lane * 4 + j;
    float val = s[id] + h[id] * g[id];
    v[j] = val; su += val; ss += val * val;
  }
  su = wave_sum(su); ss = wave_sum(ss);
  float mean = su * (1.f / 128.f);
  float var = ss * (1.f / 128.f) - mean * mean;
  float inv = rsqrtf(var + 1e-5f);
  #pragma unroll
  for (int j = 0; j < 4; ++j) {
    int i = lane * 4 + j;
    out[(size_t)n * 128 + i] = f2bf((v[j] - mean) * inv * gam[i] + bet[i]);
  }
}

// LSTM cell: gates = pre[b,t,:] + tmp[b,:] + bhh ; PyTorch order i,f,g,o
// h (bf16, feeds recurrent GEMM), c (f32), y (bf16, feeds next-layer GEMMs)
__global__ void lstm_cell_kernel(const float* __restrict__ pre, const float* __restrict__ tmp,
                                 const float* __restrict__ bhh, __bf16* __restrict__ h,
                                 float* __restrict__ c, __bf16* __restrict__ y,
                                 int B, int T, int t) {
  int idx = blockIdx.x * blockDim.x + threadIdx.x;
  if (idx >= B * 128) return;
  int b = idx >> 7, j = idx & 127;
  size_t pr = ((size_t)b * T + t) * 512;
  size_t tr = (size_t)b * 512;
  float gi = pre[pr + j]       + tmp[tr + j]       + bhh[j];
  float gf = pre[pr + 128 + j] + tmp[tr + 128 + j] + bhh[128 + j];
  float gg = pre[pr + 256 + j] + tmp[tr + 256 + j] + bhh[256 + j];
  float go = pre[pr + 384 + j] + tmp[tr + 384 + j] + bhh[384 + j];
  float cc = sigm(gf) * c[idx] + sigm(gi) * tanhf(gg);
  float hh = sigm(go) * tanhf(cc);
  c[idx] = cc;
  h[idx] = f2bf(hh);
  y[((size_t)b * T + t) * 128 + j] = f2bf(hh);
}

// dec_in[b,t,:] = enc_out[b,Te-1,:]  (bf16 -> bf16)
__global__ void repeat_last_kernel(const __bf16* __restrict__ encY, __bf16* __restrict__ decIn,
                                   int B, int Te, int Tf) {
  int idx = blockIdx.x * blockDim.x + threadIdx.x;
  if (idx >= B * Tf * 128) return;
  int j = idx & 127;
  int bt = idx >> 7;
  int b = bt / Tf;
  decIn[idx] = encY[((size_t)b * Te + (Te - 1)) * 128 + j];
}

// One (batch, head) per block. Tq=24, Tk=168, dh=32. Q/K/V fp32, O bf16.
__global__ void attn_kernel(const float* __restrict__ Q, const float* __restrict__ Kb,
                            const float* __restrict__ Vb, __bf16* __restrict__ O, int NH) {
  const int Tq = 24, Tk = 168, dh = 32;
  int b = blockIdx.x / NH, hh = blockIdx.x % NH;
  __shared__ float qs[Tq * dh];
  __shared__ float sc[Tq * Tk];
  int tid = threadIdx.x;

  for (int i = tid; i < Tq * dh; i += 256) {
    int tq = i / dh, d = i % dh;
    qs[i] = Q[((size_t)b * Tq + tq) * 128 + hh * dh + d];
  }
  __syncthreads();

  const float scale = 0.17677669529663687f;  // 1/sqrt(32)
  for (int p = tid; p < Tq * Tk; p += 256) {
    int tq = p / Tk, tk = p % Tk;
    const float* kp = Kb + ((size_t)b * Tk + tk) * 128 + hh * dh;
    float s = 0.f;
    #pragma unroll 8
    for (int d = 0; d < dh; ++d) s += qs[tq * dh + d] * kp[d];
    sc[p] = s * scale;
  }
  __syncthreads();

  int wave = tid >> 5, lane = tid & 31;
  for (int tq = wave; tq < Tq; tq += 8) {
    float mx = -1e30f;
    for (int tk = lane; tk < Tk; tk += 32) mx = fmaxf(mx, sc[tq * Tk + tk]);
    mx = wave_max(mx);
    float sum = 0.f;
    for (int tk = lane; tk < Tk; tk += 32) {
      float e = __expf(sc[tq * Tk + tk] - mx);
      sc[tq * Tk + tk] = e; sum += e;
    }
    sum = wave_sum(sum);
    float inv = 1.f / sum;
    for (int tk = lane; tk < Tk; tk += 32) sc[tq * Tk + tk] *= inv;
  }
  __syncthreads();

  for (int i = tid; i < Tq * dh; i += 256) {
    int tq = i / dh, d = i % dh;
    float o = 0.f;
    for (int tk = 0; tk < Tk; ++tk)
      o += sc[tq * Tk + tk] * Vb[((size_t)b * Tk + tk) * 128 + hh * dh + d];
    O[((size_t)b * Tq + tq) * 128 + hh * dh + d] = f2bf(o);
  }
}

// ---------------------------------------------------------------------------

extern "C" void kernel_launch(void* const* d_in, const int* in_sizes, int n_in,
                              void* d_out, int out_size, void* d_ws, size_t ws_size,
                              hipStream_t stream) {
  (void)in_sizes; (void)n_in; (void)out_size; (void)ws_size;
  const float* x         = (const float*)d_in[0];
  const float* vfc1_w    = (const float*)d_in[1];
  const float* vfc1_b    = (const float*)d_in[2];
  const float* vfc2_w    = (const float*)d_in[3];
  const float* vfc2_b    = (const float*)d_in[4];
  const float* vgate_w   = (const float*)d_in[5];
  const float* vgate_b   = (const float*)d_in[6];
  const float* vskip_w   = (const float*)d_in[7];
  const float* vskip_b   = (const float*)d_in[8];
  const float* vln_g     = (const float*)d_in[9];
  const float* vln_b     = (const float*)d_in[10];
  const float* gfc1_w    = (const float*)d_in[11];
  const float* gfc1_b    = (const float*)d_in[12];
  const float* gfc2_w    = (const float*)d_in[13];
  const float* gfc2_b    = (const float*)d_in[14];
  const float* ggate_w   = (const float*)d_in[15];
  const float* ggate_b   = (const float*)d_in[16];
  const float* gskip_w   = (const float*)d_in[17];
  const float* gskip_b   = (const float*)d_in[18];
  const float* gln_g     = (const float*)d_in[19];
  const float* gln_b     = (const float*)d_in[20];
  const float* enc_wih   = (const float*)d_in[21];
  const float* enc_whh   = (const float*)d_in[22];
  const float* enc_bih   = (const float*)d_in[23];
  const float* enc_bhh   = (const float*)d_in[24];
  const float* dec_wih   = (const float*)d_in[25];
  const float* dec_whh   = (const float*)d_in[26];
  const float* dec_bih   = (const float*)d_in[27];
  const float* dec_bhh   = (const float*)d_in[28];
  const float* ain_w     = (const float*)d_in[29];
  const float* ain_b     = (const float*)d_in[30];
  const float* aout_w    = (const float*)d_in[31];
  const float* aout_b    = (const float*)d_in[32];
  const float* pfc1_w    = (const float*)d_in[33];
  const float* pfc1_b    = (const float*)d_in[34];
  const float* pfc2_w    = (const float*)d_in[35];
  const float* pfc2_b    = (const float*)d_in[36];
  const float* pgate_w   = (const float*)d_in[37];
  const float* pgate_b   = (const float*)d_in[38];
  const float* pln_g     = (const float*)d_in[39];
  const float* pln_b     = (const float*)d_in[40];
  const float* fo_w      = (const float*)d_in[41];
  const float* fo_b      = (const float*)d_in[42];

  const int B = 256, Te = 168, Tf = 24;
  const size_t Ne = (size_t)B * Te;   // 43008 encoder rows
  const size_t Nd = (size_t)B * Tf;   // 6144 decoder rows

  // ---- workspace layout (byte allocator, 256B aligned) ----
  char* base = (char*)d_ws;
  size_t off = 0;
  auto allocf = [&](size_t n) { float* p = (float*)(base + off);
                                off = (off + n * 4 + 255) & ~(size_t)255; return p; };
  auto allocb = [&](size_t n) { __bf16* p = (__bf16*)(base + off);
                                off = (off + n * 2 + 255) & ~(size_t)255; return p; };

  __bf16* procB  = allocb(Ne * 1152);  // VSN proc (bf16); fp32 pre reuses these bytes
  __bf16* h1b    = allocb(Ne * 128);   // VSN h1 / gc_h1
  __bf16* encInb = allocb(Ne * 128);   // sel output = LSTM layer0 input
  __bf16* Y0b    = allocb(Ne * 128);   // encoder layer0 output
  __bf16* encYb  = allocb(Ne * 128);   // encoder output
  float*  tBf    = allocf(Ne * 128);   // vsn/gc h (f32); reused as V
  float*  tCf    = allocf(Ne * 128);   // vsn/gc g (f32); reused as K
  float*  tDf    = allocf(Ne * 128);   // gc skip (f32)
  float*  tmp    = allocf((size_t)B * 512);
  float*  Cst    = allocf((size_t)2 * B * 128);       // c states (L0, L1)
  __bf16* Hb     = allocb((size_t)2 * B * 128);       // h states (L0, L1), bf16
  __bf16* decInb = allocb(Nd * 128);
  float*  preD   = allocf(Nd * 512);
  __bf16* Yd0b   = allocb(Nd * 128);
  __bf16* Yd1b   = allocb(Nd * 128);
  float*  Qf     = allocf(Nd * 128);
  __bf16* Obb    = allocb(Nd * 128);
  float*  attnBf = allocf(Nd * 128);
  __bf16* attnBb = allocb(Nd * 128);
  __bf16* p1b    = allocb(Nd * 128);
  float*  p2f    = allocf(Nd * 128);
  float*  p3f    = allocf(Nd * 128);
  __bf16* paOb   = allocb(Nd * 128);
  // bf16 weights
  __bf16* wb_vfc2  = allocb((size_t)9 * 128 * 128);
  __bf16* wb_vgate = allocb((size_t)9 * 128 * 128);
  __bf16* wb_gfc1  = allocb((size_t)128 * 1152);
  __bf16* wb_gskip = allocb((size_t)128 * 1152);
  __bf16* wb_gfc2  = allocb(128 * 128);
  __bf16* wb_ggate = allocb(128 * 128);
  __bf16* wb_ewih  = allocb((size_t)2 * 512 * 128);
  __bf16* wb_ewhh  = allocb((size_t)2 * 512 * 128);
  __bf16* wb_dwih  = allocb((size_t)2 * 512 * 128);
  __bf16* wb_dwhh  = allocb((size_t)2 * 512 * 128);
  __bf16* wb_ain   = allocb(384 * 128);
  __bf16* wb_aout  = allocb(128 * 128);
  __bf16* wb_pfc1  = allocb(128 * 128);
  __bf16* wb_pfc2  = allocb(128 * 128);
  __bf16* wb_pgate = allocb(128 * 128);
  __bf16* wb_fo    = allocb(6 * 128);

  float* preE = (float*)procB;  // reuse: proc (bf16 99MB) dead after gc GEMMs; pre f32 88MB
  float* Kf = tCf; float* Vf = tBf;  // reuse: dead after sel
  __bf16* Hb0 = Hb; __bf16* Hb1 = Hb + (size_t)B * 128;
  float* C0s = Cst; float* C1s = Cst + (size_t)B * 128;

  const int EW = 256;

  // ---- one-time (per launch) weight conversions ----
  conv(stream, vfc2_w, wb_vfc2, 9 * 128 * 128);
  conv(stream, vgate_w, wb_vgate, 9 * 128 * 128);
  conv(stream, gfc1_w, wb_gfc1, 128 * 1152);
  conv(stream, gskip_w, wb_gskip, 128 * 1152);
  conv(stream, gfc2_w, wb_gfc2, 128 * 128);
  conv(stream, ggate_w, wb_ggate, 128 * 128);
  conv(stream, enc_wih, wb_ewih, 2 * 512 * 128);
  conv(stream, enc_whh, wb_ewhh, 2 * 512 * 128);
  conv(stream, dec_wih, wb_dwih, 2 * 512 * 128);
  conv(stream, dec_whh, wb_dwhh, 2 * 512 * 128);
  conv(stream, ain_w, wb_ain, 384 * 128);
  conv(stream, aout_w, wb_aout, 128 * 128);
  conv(stream, pfc1_w, wb_pfc1, 128 * 128);
  conv(stream, pfc2_w, wb_pfc2, 128 * 128);
  conv(stream, pgate_w, wb_pgate, 128 * 128);
  conv(stream, fo_w, wb_fo, 6 * 128);

  // ---- VSN: per-feature GRNs (input dim 1) ----
  for (int f = 0; f < 9; ++f) {
    vsn_h1_kernel<<<(unsigned)((Ne * 128 + EW - 1) / EW), EW, 0, stream>>>(x, vfc1_w, vfc1_b, h1b, (int)Ne, f);
    gemm(stream, h1b, 128, wb_vfc2 + (size_t)f * 128 * 128, 128, vfc2_b + f * 128, tBf, nullptr, 128, (int)Ne, 128, 128, 0);
    gemm(stream, h1b, 128, wb_vgate + (size_t)f * 128 * 128, 128, vgate_b + f * 128, tCf, nullptr, 128, (int)Ne, 128, 128, 2);
    vsn_proc_kernel<<<(unsigned)((Ne + 7) / 8), EW, 0, stream>>>(x, vskip_w, vskip_b, tBf, tCf, vln_g, vln_b,
                                                                 procB + (size_t)f * 128, 1152, (int)Ne, f);
  }

  // ---- gc GRN over flattened [Ne,1152] -> sel [Ne,128] ----
  gemm(stream, procB, 1152, wb_gfc1, 1152, gfc1_b, nullptr, h1b, 128, (int)Ne, 128, 1152, 1);  // elu fused
  gemm(stream, procB, 1152, wb_gskip, 1152, gskip_b, tDf, nullptr, 128, (int)Ne, 128, 1152, 0);
  gemm(stream, h1b, 128, wb_gfc2, 128, gfc2_b, tBf, nullptr, 128, (int)Ne, 128, 128, 0);
  gemm(stream, h1b, 128, wb_ggate, 128, ggate_b, tCf, nullptr, 128, (int)Ne, 128, 128, 2);     // sigmoid fused
  grn_finish_kernel<<<(unsigned)((Ne + 7) / 8), EW, 0, stream>>>(tDf, tBf, tCf, gln_g, gln_b, encInb, (int)Ne);

  // ---- encoder LSTM (2 layers) ----
  fill_zero_kernel<<<(2 * B * 128 + EW - 1) / EW, EW, 0, stream>>>(Cst, 2 * B * 128);
  fill_zero_kernel<<<(B * 128 + EW - 1) / EW, EW, 0, stream>>>((float*)Hb, B * 128);  // 2*B*128 bf16
  gemm(stream, encInb, 128, wb_ewih, 128, enc_bih, preE, nullptr, 512, (int)Ne, 512, 128, 0);
  for (int t = 0; t < Te; ++t) {
    gemm(stream, Hb0, 128, wb_ewhh, 128, nullptr, tmp, nullptr, 512, B, 512, 128, 0);
    lstm_cell_kernel<<<(B * 128 + EW - 1) / EW, EW, 0, stream>>>(preE, tmp, enc_bhh, Hb0, C0s, Y0b, B, Te, t);
  }
  gemm(stream, Y0b, 128, wb_ewih + 512 * 128, 128, enc_bih + 512, preE, nullptr, 512, (int)Ne, 512, 128, 0);
  for (int t = 0; t < Te; ++t) {
    gemm(stream, Hb1, 128, wb_ewhh + 512 * 128, 128, nullptr, tmp, nullptr, 512, B, 512, 128, 0);
    lstm_cell_kernel<<<(B * 128 + EW - 1) / EW, EW, 0, stream>>>(preE, tmp, enc_bhh + 512, Hb1, C1s, encYb, B, Te, t);
  }

  // ---- decoder LSTM (states continue from encoder finals) ----
  repeat_last_kernel<<<(unsigned)((Nd * 128 + EW - 1) / EW), EW, 0, stream>>>(encYb, decInb, B, Te, Tf);
  gemm(stream, decInb, 128, wb_dwih, 128, dec_bih, preD, nullptr, 512, (int)Nd, 512, 128, 0);
  for (int t = 0; t < Tf; ++t) {
    gemm(stream, Hb0, 128, wb_dwhh, 128, nullptr, tmp, nullptr, 512, B, 512, 128, 0);
    lstm_cell_kernel<<<(B * 128 + EW - 1) / EW, EW, 0, stream>>>(preD, tmp, dec_bhh, Hb0, C0s, Yd0b, B, Tf, t);
  }
  gemm(stream, Yd0b, 128, wb_dwih + 512 * 128, 128, dec_bih + 512, preD, nullptr, 512, (int)Nd, 512, 128, 0);
  for (int t = 0; t < Tf; ++t) {
    gemm(stream, Hb1, 128, wb_dwhh + 512 * 128, 128, nullptr, tmp, nullptr, 512, B, 512, 128, 0);
    lstm_cell_kernel<<<(B * 128 + EW - 1) / EW, EW, 0, stream>>>(preD, tmp, dec_bhh + 512, Hb1, C1s, Yd1b, B, Tf, t);
  }

  // ---- cross attention (decoder -> encoder) ----
  gemm(stream, Yd1b, 128, wb_ain, 128, ain_b, Qf, nullptr, 128, (int)Nd, 128, 128, 0);
  gemm(stream, encYb, 128, wb_ain + 128 * 128, 128, ain_b + 128, Kf, nullptr, 128, (int)Ne, 128, 128, 0);
  gemm(stream, encYb, 128, wb_ain + 256 * 128, 128, ain_b + 256, Vf, nullptr, 128, (int)Ne, 128, 128, 0);
  attn_kernel<<<B * 4, 256, 0, stream>>>(Qf, Kf, Vf, Obb, 4);
  gemm(stream, Obb, 128, wb_aout, 128, aout_b, attnBf, attnBb, 128, (int)Nd, 128, 128, 0);

  // ---- pa GRN (identity skip) + final projection ----
  gemm(stream, attnBb, 128, wb_pfc1, 128, pfc1_b, nullptr, p1b, 128, (int)Nd, 128, 128, 1);
  gemm(stream, p1b, 128, wb_pfc2, 128, pfc2_b, p2f, nullptr, 128, (int)Nd, 128, 128, 0);
  gemm(stream, p1b, 128, wb_pgate, 128, pgate_b, p3f, nullptr, 128, (int)Nd, 128, 128, 2);
  grn_finish_kernel<<<(unsigned)((Nd + 7) / 8), EW, 0, stream>>>(attnBf, p2f, p3f, pln_g, pln_b, paOb, (int)Nd);
  gemm(stream, paOb, 128, wb_fo, 128, fo_b, (float*)d_out, nullptr, 6, (int)Nd, 6, 128, 0);
}
